// MixtureOfExperts_49108656063220
// MI455X (gfx1250) — matrix operand order repaired and checked by reading, not compiled, not run on hardware
//
#include <hip/hip_runtime.h>

// MoE top-1 (Switch-style) layer for gfx1250 / MI455X.
// T=8192 tokens, M=1024, E=8, DFF=4096, C=1280.
// Expert FFN GEMMs run on v_wmma_f32_16x16x32_bf16; tiles are DMA'd into LDS
// by the Tensor Data Mover (tensor_load_to_lds + s_wait_tensorcnt),
// double-buffered, so compute waves never issue global loads.

#define TOKS 8192
#define MDIM 1024
#define NEXP 8
#define DFF  4096
#define CAPC 1280

typedef __attribute__((ext_vector_type(16))) __bf16 v16bf;
typedef __attribute__((ext_vector_type(8)))  __bf16 v8bf;
typedef __attribute__((ext_vector_type(8)))  float  v8f;

#if defined(__gfx1250__) && __has_builtin(__builtin_amdgcn_tensor_load_to_lds) && \
    __has_builtin(__builtin_amdgcn_s_wait_tensorcnt)
#define HAS_TDM 1
#else
#define HAS_TDM 0
#endif

#if HAS_TDM
typedef unsigned int u32x4 __attribute__((ext_vector_type(4)));
typedef int          i32x4 __attribute__((ext_vector_type(4)));
typedef int          i32x8 __attribute__((ext_vector_type(8)));

// 2D tile load: 32 contiguous bf16 x 128 rows, row pitch = row_len elements.
// LDS rows padded +16B every 64B (pad_interval=3 -> 16 DWORDs, pad_amount=3 -> 4 DWORDs)
// giving an 80B LDS row pitch (LDA/LDB = 40 bf16).
__device__ __forceinline__ void tdm_tile_load(const __bf16* gsrc, const __bf16* lds_dst,
                                              unsigned row_len_elems) {
  const unsigned long ga = (unsigned long)gsrc;
  const unsigned lo = (unsigned)(unsigned long)lds_dst;   // addr[31:0] = LDS offset
  u32x4 g0; i32x8 g1; i32x4 g2 = {}; i32x4 g3 = {};
  g0[0] = 1u;                                             // count=1, user descriptor
  g0[1] = lo;                                             // lds_addr
  g0[2] = (unsigned)ga;                                   // global_addr[31:0]
  g0[3] = (unsigned)((ga >> 32) & 0x01FFFFFFu) | (2u << 30);  // global_addr[56:32], type=2
  // data_size=2B, pad_enable, pad_interval=3 (16 DW), pad_amount=3 (4 DW)
  g1[0] = (int)((1u << 16) | (1u << 20) | (3u << 22) | (3u << 25));
  g1[1] = (int)((32u & 0xFFFFu) << 16);                   // tensor_dim0 = 32 (lo16)
  g1[2] = (int)((128u & 0xFFFFu) << 16);                  // dim0 hi=0 | tensor_dim1 = 128
  g1[3] = (int)(32u << 16);                               // dim1 hi=0 | tile_dim0 = 32
  g1[4] = 128;                                            // tile_dim1 = 128, tile_dim2 = 0
  g1[5] = (int)row_len_elems;                             // tensor_dim0_stride lo32
  g1[6] = 0;
  g1[7] = 0;
#if __has_include(<hip/amd_detail/amd_gfx1250_TDM.h>)
  i32x8 z = {};
  __builtin_amdgcn_tensor_load_to_lds(g0, g1, g2, g3, z, 0);
#else
  __builtin_amdgcn_tensor_load_to_lds(g0, g1, g2, g3, 0);
#endif
}
#endif  // HAS_TDM

__device__ __forceinline__ __bf16 f2bf(float f) {
  union { float f; unsigned u; } in; in.f = f;
  unsigned u = in.u;
  unsigned r = (u + 0x7FFFu + ((u >> 16) & 1u)) >> 16;   // round-to-nearest-even
  union { unsigned short s; __bf16 b; } out; out.s = (unsigned short)r;
  return out.b;
}

// ---------------------------------------------------------------- init
__global__ void moe_init(float* __restrict__ out, int* __restrict__ slot_token,
                         float* __restrict__ slot_gate) {
  long i = (long)blockIdx.x * blockDim.x + threadIdx.x;
  if (i < (long)TOKS * MDIM) out[i] = 0.0f;
  if (i < NEXP * CAPC) { slot_token[i] = -1; slot_gate[i] = 0.0f; }
}

// -------------------------------------------------- transpose + f32->bf16
// Per expert: src [R][Cc] f32 -> dst [Cc][R] bf16, 32x32 tiles through LDS.
__global__ __launch_bounds__(256)
void moe_cvtT(const float* __restrict__ src, __bf16* __restrict__ dst, int R, int Cc) {
  __shared__ float t[32][33];
  const int tx = threadIdx.x & 31, ty = threadIdx.x >> 5;
  const int e = blockIdx.z;
  const int c0 = blockIdx.x * 32;
  const int r0 = blockIdx.y * 32;
  const float* s = src + (long)e * R * Cc;
  __bf16* d = dst + (long)e * Cc * R;
#pragma unroll
  for (int i = 0; i < 4; i++)
    t[ty + 8 * i][tx] = s[(long)(r0 + ty + 8 * i) * Cc + c0 + tx];
  __syncthreads();
#pragma unroll
  for (int i = 0; i < 4; i++)
    d[(long)(c0 + ty + 8 * i) * R + r0 + tx] = f2bf(t[tx][ty + 8 * i]);
}

// ---------------------------------------------------------------- gating
__global__ __launch_bounds__(256)
void moe_gate(const float* __restrict__ x, const float* __restrict__ wg,
              int* __restrict__ eidx, int* __restrict__ rankv,
              float* __restrict__ prob, int* __restrict__ chunkCnt) {
  __shared__ float wgs[MDIM * NEXP];      // 32 KB
  __shared__ int   se[256];
  const int tid = threadIdx.x;
  for (int i = tid; i < MDIM * NEXP; i += 256) wgs[i] = wg[i];
  __syncthreads();

  const int t = blockIdx.x * 256 + tid;
  const float* xr = x + (long)t * MDIM;
  float l[NEXP];
#pragma unroll
  for (int e = 0; e < NEXP; e++) l[e] = 0.0f;
  for (int m = 0; m < MDIM; m++) {
    float xv = xr[m];
#pragma unroll
    for (int e = 0; e < NEXP; e++) l[e] = fmaf(xv, wgs[m * NEXP + e], l[e]);
  }
  float mx = l[0];
#pragma unroll
  for (int e = 1; e < NEXP; e++) mx = fmaxf(mx, l[e]);
  float s = 0.0f, ex[NEXP];
#pragma unroll
  for (int e = 0; e < NEXP; e++) { ex[e] = __expf(l[e] - mx); s += ex[e]; }
  int best = 0; float bl = l[0];
#pragma unroll
  for (int e = 1; e < NEXP; e++) if (l[e] > bl) { bl = l[e]; best = e; }

  prob[t] = ex[best] / s;
  eidx[t] = best;
  se[tid] = best;
  __syncthreads();

  int r = 0;
  for (int j = 0; j < tid; j++) r += (se[j] == best);
  rankv[t] = r;
  if (tid < NEXP) {
    int c = 0;
    for (int j = 0; j < 256; j++) c += (se[j] == tid);
    chunkCnt[blockIdx.x * NEXP + tid] = c;
  }
}

// ---------------------------------------------------------------- chunk scan
__global__ void moe_scan(const int* __restrict__ chunkCnt, int* __restrict__ chunkOff) {
  int e = threadIdx.x;
  if (e < NEXP) {
    int run = 0;
    for (int b = 0; b < TOKS / 256; b++) {
      chunkOff[b * NEXP + e] = run;
      run += chunkCnt[b * NEXP + e];
    }
  }
}

// ---------------------------------------------------------------- dispatch
__global__ __launch_bounds__(256)
void moe_dispatch(const int* __restrict__ eidx, const int* __restrict__ rankv,
                  const float* __restrict__ prob, const int* __restrict__ chunkOff,
                  int* __restrict__ slot_token, float* __restrict__ slot_gate) {
  const int t = blockIdx.x * 256 + threadIdx.x;
  const int e = eidx[t];
  const int pos = chunkOff[blockIdx.x * NEXP + e] + rankv[t];
  if (pos < CAPC) {
    slot_token[e * CAPC + pos] = t;
    slot_gate[e * CAPC + pos]  = prob[t];
  }
}

// ------------------------------------------- gather tokens -> bf16 slot rows
__global__ __launch_bounds__(256)
void moe_gather(const float* __restrict__ x, const int* __restrict__ slot_token,
                __bf16* __restrict__ xd) {
  const int slot = blockIdx.x;                 // 0 .. E*C-1
  const int tok = slot_token[slot];
  __bf16* d = xd + (long)slot * MDIM;
  const int i0 = threadIdx.x * 4;
  if (tok < 0) {
    const __bf16 z = f2bf(0.0f);
    d[i0] = z; d[i0 + 1] = z; d[i0 + 2] = z; d[i0 + 3] = z;
  } else {
    const float4 v = *(const float4*)(x + (long)tok * MDIM + i0);
    d[i0] = f2bf(v.x); d[i0 + 1] = f2bf(v.y);
    d[i0 + 2] = f2bf(v.z); d[i0 + 3] = f2bf(v.w);
  }
}

// ---------------------------------------------------------------- GEMMs
#define LDA 40   // bf16 per LDS row: 64B data + 16B pad (matches TDM pad descriptor)
#define LDB 40

__device__ __forceinline__ v16bf pack16(v8bf lo, v8bf hi) {
  v16bf f;
#pragma unroll
  for (int i = 0; i < 8; i++) { f[i] = lo[i]; f[i + 8] = hi[i]; }
  return f;
}

// GEMM1: h[e] = relu(xd[e] @ w1[e] + b1[e])   A: xd [C][M], B: w1T [DFF][M] (n-major)
__global__ __launch_bounds__(256)
void moe_ffn1(const __bf16* __restrict__ xd, const __bf16* __restrict__ w1t,
              const float* __restrict__ b1, __bf16* __restrict__ h) {
  __shared__ __align__(16) __bf16 As[2][128 * LDA];
  __shared__ __align__(16) __bf16 Bs[2][128 * LDB];

  const int tid  = threadIdx.x;
  const int wave = tid >> 5, lane = tid & 31;
  const int half = lane >> 4, lr = lane & 15;
  const int e = blockIdx.z;
  const int rowBase = blockIdx.y * 128;
  const int n0 = blockIdx.x * 128;
  const int wr = (wave >> 2) * 64, wc = (wave & 3) * 32;

  const __bf16* aT = xd  + ((long)e * CAPC + rowBase) * MDIM;
  const __bf16* bT = w1t + ((long)e * DFF  + n0)      * MDIM;

  v8f acc[4][2] = {};
  int cur = 0;

#if HAS_TDM
  if (wave == 0) {
    tdm_tile_load(aT, &As[0][0], MDIM);
    tdm_tile_load(bT, &Bs[0][0], MDIM);
  }
#else
  const int ar  = tid >> 1;
  const int ako = (tid & 1) * 16;
#endif

  for (int k0 = 0; k0 < MDIM; k0 += 32) {
#if HAS_TDM
    if (wave == 0) __builtin_amdgcn_s_wait_tensorcnt(0);
    __syncthreads();
    if (k0 + 32 < MDIM && wave == 0) {
      tdm_tile_load(aT + k0 + 32, &As[cur ^ 1][0], MDIM);
      tdm_tile_load(bT + k0 + 32, &Bs[cur ^ 1][0], MDIM);
    }
#else
    __syncthreads();
    {
      const v8bf* s = (const v8bf*)(aT + (long)ar * MDIM + k0 + ako);
      __bf16* d = &As[0][ar * LDA + ako];
      v8bf a0 = s[0], a1 = s[1];
#pragma unroll
      for (int i = 0; i < 8; i++) { d[i] = a0[i]; d[i + 8] = a1[i]; }
    }
    {
      const v8bf* s = (const v8bf*)(bT + (long)ar * MDIM + k0 + ako);
      __bf16* d = &Bs[0][ar * LDB + ako];
      v8bf a0 = s[0], a1 = s[1];
#pragma unroll
      for (int i = 0; i < 8; i++) { d[i] = a0[i]; d[i + 8] = a1[i]; }
    }
    __syncthreads();
#endif

    v16bf bfr[2];
#pragma unroll
    for (int nt = 0; nt < 2; nt++) {
      const __bf16* bp = &Bs[cur][(wc + nt * 16 + lr) * LDB + half * 16];
      bfr[nt] = pack16(*(const v8bf*)bp, *(const v8bf*)(bp + 8));
    }
#pragma unroll
    for (int mt = 0; mt < 4; mt++) {
      const __bf16* ap = &As[cur][(wr + mt * 16 + lr) * LDA + half * 16];
      v16bf a = pack16(*(const v8bf*)ap, *(const v8bf*)(ap + 8));
      acc[mt][0] = __builtin_amdgcn_wmma_f32_16x16x32_bf16(
          false, a, false, bfr[0], (short)0, acc[mt][0], false, false);
      acc[mt][1] = __builtin_amdgcn_wmma_f32_16x16x32_bf16(
          false, a, false, bfr[1], (short)0, acc[mt][1], false, false);
    }
#if HAS_TDM
    cur ^= 1;
#endif
  }

  const long hOff = ((long)e * CAPC + rowBase) * DFF;
#pragma unroll
  for (int nt = 0; nt < 2; nt++) {
    const int gc = n0 + wc + nt * 16 + lr;
    const float bias = b1[e * DFF + gc];
#pragma unroll
    for (int mt = 0; mt < 4; mt++) {
#pragma unroll
      for (int r = 0; r < 8; r++) {
        const int rr = wr + mt * 16 + half * 8 + r;
        h[hOff + (long)rr * DFF + gc] = f2bf(fmaxf(acc[mt][nt][r] + bias, 0.0f));
      }
    }
  }
}

// GEMM2: out[token] = gate * (h[e] @ w2[e] + b2[e])  A: h [C][DFF], B: w2T [M][DFF]
__global__ __launch_bounds__(256)
void moe_ffn2(const __bf16* __restrict__ h, const __bf16* __restrict__ w2t,
              const float* __restrict__ b2, const int* __restrict__ slot_token,
              const float* __restrict__ slot_gate, float* __restrict__ out) {
  __shared__ __align__(16) __bf16 As[2][128 * LDA];
  __shared__ __align__(16) __bf16 Bs[2][128 * LDB];

  const int tid  = threadIdx.x;
  const int wave = tid >> 5, lane = tid & 31;
  const int half = lane >> 4, lr = lane & 15;
  const int e = blockIdx.z;
  const int rowBase = blockIdx.y * 128;
  const int n0 = blockIdx.x * 128;
  const int wr = (wave >> 2) * 64, wc = (wave & 3) * 32;

  const __bf16* aT = h   + ((long)e * CAPC + rowBase) * DFF;
  const __bf16* bT = w2t + ((long)e * MDIM + n0)      * DFF;

  v8f acc[4][2] = {};
  int cur = 0;

#if HAS_TDM
  if (wave == 0) {
    tdm_tile_load(aT, &As[0][0], DFF);
    tdm_tile_load(bT, &Bs[0][0], DFF);
  }
#else
  const int ar  = tid >> 1;
  const int ako = (tid & 1) * 16;
#endif

  for (int k0 = 0; k0 < DFF; k0 += 32) {
#if HAS_TDM
    if (wave == 0) __builtin_amdgcn_s_wait_tensorcnt(0);
    __syncthreads();
    if (k0 + 32 < DFF && wave == 0) {
      tdm_tile_load(aT + k0 + 32, &As[cur ^ 1][0], DFF);
      tdm_tile_load(bT + k0 + 32, &Bs[cur ^ 1][0], DFF);
    }
#else
    __syncthreads();
    {
      const v8bf* s = (const v8bf*)(aT + (long)ar * DFF + k0 + ako);
      __bf16* d = &As[0][ar * LDA + ako];
      v8bf a0 = s[0], a1 = s[1];
#pragma unroll
      for (int i = 0; i < 8; i++) { d[i] = a0[i]; d[i + 8] = a1[i]; }
    }
    {
      const v8bf* s = (const v8bf*)(bT + (long)ar * DFF + k0 + ako);
      __bf16* d = &Bs[0][ar * LDB + ako];
      v8bf a0 = s[0], a1 = s[1];
#pragma unroll
      for (int i = 0; i < 8; i++) { d[i] = a0[i]; d[i + 8] = a1[i]; }
    }
    __syncthreads();
#endif

    v16bf bfr[2];
#pragma unroll
    for (int nt = 0; nt < 2; nt++) {
      const __bf16* bp = &Bs[cur][(wc + nt * 16 + lr) * LDB + half * 16];
      bfr[nt] = pack16(*(const v8bf*)bp, *(const v8bf*)(bp + 8));
    }
#pragma unroll
    for (int mt = 0; mt < 4; mt++) {
      const __bf16* ap = &As[cur][(wr + mt * 16 + lr) * LDA + half * 16];
      v16bf a = pack16(*(const v8bf*)ap, *(const v8bf*)(ap + 8));
      acc[mt][0] = __builtin_amdgcn_wmma_f32_16x16x32_bf16(
          false, a, false, bfr[0], (short)0, acc[mt][0], false, false);
      acc[mt][1] = __builtin_amdgcn_wmma_f32_16x16x32_bf16(
          false, a, false, bfr[1], (short)0, acc[mt][1], false, false);
    }
#if HAS_TDM
    cur ^= 1;
#endif
  }

#pragma unroll
  for (int nt = 0; nt < 2; nt++) {
    const int gc = n0 + wc + nt * 16 + lr;
    const float bias = b2[e * MDIM + gc];
#pragma unroll
    for (int mt = 0; mt < 4; mt++) {
#pragma unroll
      for (int r = 0; r < 8; r++) {
        const int rr = rowBase + wr + mt * 16 + half * 8 + r;
        const int tok = slot_token[e * CAPC + rr];
        if (tok >= 0) {
          out[(long)tok * MDIM + gc] = slot_gate[e * CAPC + rr] * (acc[mt][nt][r] + bias);
        }
      }
    }
  }
}

// ---------------------------------------------------------------- launcher
extern "C" void kernel_launch(void* const* d_in, const int* in_sizes, int n_in,
                              void* d_out, int out_size, void* d_ws, size_t ws_size,
                              hipStream_t stream) {
  const float* x  = (const float*)d_in[0];
  const float* wg = (const float*)d_in[1];
  const float* w1 = (const float*)d_in[2];
  const float* b1 = (const float*)d_in[3];
  const float* w2 = (const float*)d_in[4];
  const float* b2 = (const float*)d_in[5];
  float* out = (float*)d_out;

  char* ws = (char*)d_ws;
  int*   eidx       = (int*)  (ws + 0);
  int*   rankv      = (int*)  (ws + 32768);
  float* prob       = (float*)(ws + 65536);
  int*   chunkCnt   = (int*)  (ws + 98304);
  int*   chunkOff   = (int*)  (ws + 99328);
  int*   slot_token = (int*)  (ws + 100352);
  float* slot_gate  = (float*)(ws + 141312);
  __bf16* w1t  = (__bf16*)(ws + 262144L);
  __bf16* w2t  = (__bf16*)(ws + 262144L + 67108864L);
  __bf16* hbuf = (__bf16*)(ws + 262144L + 2 * 67108864L);                 // 80 MB
  __bf16* xd   = (__bf16*)(ws + 262144L + 2 * 67108864L + 83886080L);     // 20 MB

  const long nOut = (long)TOKS * MDIM;
  moe_init<<<(unsigned)((nOut + 255) / 256), 256, 0, stream>>>(out, slot_token, slot_gate);

  // weights: transpose + convert to bf16 (one-time, n-major for straight tile copies)
  dim3 gc1(DFF / 32, MDIM / 32, NEXP);
  moe_cvtT<<<gc1, 256, 0, stream>>>(w1, w1t, MDIM, DFF);
  dim3 gc2(MDIM / 32, DFF / 32, NEXP);
  moe_cvtT<<<gc2, 256, 0, stream>>>(w2, w2t, DFF, MDIM);

  moe_gate<<<TOKS / 256, 256, 0, stream>>>(x, wg, eidx, rankv, prob, chunkCnt);
  moe_scan<<<1, 32, 0, stream>>>(chunkCnt, chunkOff);
  moe_dispatch<<<TOKS / 256, 256, 0, stream>>>(eidx, rankv, prob, chunkOff,
                                               slot_token, slot_gate);
  moe_gather<<<NEXP * CAPC, 256, 0, stream>>>(x, slot_token, xd);

  dim3 g1(DFF / 128, CAPC / 128, NEXP);   // 32 x 10 x 8
  moe_ffn1<<<g1, 256, 0, stream>>>(xd, w1t, b1, hbuf);

  dim3 g2(MDIM / 128, CAPC / 128, NEXP);  // 8 x 10 x 8
  moe_ffn2<<<g2, 256, 0, stream>>>(hbuf, w2t, b2, slot_token, slot_gate, out);
}
